// AttentionHead_13572096655919
// MI455X (gfx1250) — matrix operand order repaired and checked by reading, not compiled, not run on hardware
//
#include <hip/hip_runtime.h>
#include <hip/hip_bf16.h>

typedef __attribute__((ext_vector_type(16))) _Float16 v16h;
typedef __attribute__((ext_vector_type(8)))  _Float16 v8h;
typedef __attribute__((ext_vector_type(8)))  float    v8f;
typedef __attribute__((ext_vector_type(4)))  float    v4f;

namespace {
constexpr int   kB     = 16;
constexpr int   kT     = 2048;
constexpr int   kDin   = 1024;
constexpr int   kH     = 64;
constexpr float kDropP = 0.2f;
constexpr float kKeep  = 1.0f / (1.0f - kDropP);
constexpr float kSm    = 0.125f;   // HEAD^-0.5
// Wp: repacked weights, per-lane WMMA B-operand image:
// [kchunk(32)][mat*4+ntile(12)][lane(32)][elem(16)] f16
constexpr int kWpChunkStride = 12 * 32 * 16;   // elems per K-chunk
constexpr int kWpMatStride   = 4 * 32 * 16;    // elems per matrix within chunk
}

union HV32 { v16h v; v8h h[2]; };

static __device__ __forceinline__ v8f wmma16(v16h a, v16h b, v8f c) {
  return __builtin_amdgcn_wmma_f32_16x16x32_f16(false, a, false, b, (short)0, c,
                                                false, false);
}

// LDS aperture low 32 bits == LDS byte offset (ISA 10.2), so truncating the
// generic pointer to a __shared__ object yields the VDST byte address for
// async-to-LDS ops.
static __device__ __forceinline__ unsigned int lds_byte_off(const void* p) {
  return (unsigned int)(unsigned long long)p;
}

// CDNA5 async copy global->LDS (ASYNCcnt-tracked), 16B per lane.
static __device__ __forceinline__ void async_load_b128(unsigned int lds_off,
                                                       const void* gaddr) {
  unsigned long long ga = (unsigned long long)gaddr;
  asm volatile("global_load_async_to_lds_b128 %0, %1, off"
               :: "v"(lds_off), "v"(ga)
               : "memory");
}

static __device__ __forceinline__ void wait_asynccnt0() {
#if __has_builtin(__builtin_amdgcn_s_wait_asynccnt)
  __builtin_amdgcn_s_wait_asynccnt(0);
#else
  asm volatile("s_wait_asynccnt 0x0" ::: "memory");
#endif
}

// -------------------------------------------------------------------------
// Kernel 0: repack Wq/Wk/Wv (f32, [1024][64]) into the exact per-lane f16
// WMMA B-operand register image so kernel 1 loads B with one 32B load.
// grid = (32 kchunks, 12 mat*ntile), block = 32
// -------------------------------------------------------------------------
__global__ __launch_bounds__(32) void w_repack_kernel(
    const float* __restrict__ Wq, const float* __restrict__ Wk,
    const float* __restrict__ Wv, _Float16* __restrict__ Wp) {
  const int lane = threadIdx.x & 31;
  const int ml   = lane & 15;
  const int hh   = lane >> 4;
  const int c    = blockIdx.x;          // K chunk (32 wide)
  const int mn   = blockIdx.y;          // mat*4 + ntile
  const int mat  = mn >> 2;
  const int nt   = mn & 3;

  const float* W = (mat == 0) ? Wq : (mat == 1) ? Wk : Wv;
  _Float16* dst  = Wp + ((size_t)(c * 12 + mn) * 32 + lane) * 16;
#pragma unroll
  for (int e = 0; e < 16; ++e)   // B layout: elem e <-> K = hh*8 + e (e<8), 16 + hh*8 + (e-8) (e>=8)
    dst[e] = (_Float16)W[(size_t)(c * 32 + (e < 8 ? hh * 8 + e : 16 + hh * 8 + (e - 8))) * kH + nt * 16 + ml];
}

// -------------------------------------------------------------------------
// Kernel 1: q/k/v projections fused. One block = 16 rows of x; the x tile is
// async-copied to LDS once and shared by 4 waves x 3 matrices.
// grid = (B*T/16), block = 128 (wave w owns N-tile w of q, k and vT)
// -------------------------------------------------------------------------
__global__ __launch_bounds__(128) void qkv_proj_kernel(
    const float* __restrict__ x, const _Float16* __restrict__ Wp,
    _Float16* __restrict__ q_out, _Float16* __restrict__ k_out,
    _Float16* __restrict__ vt_out) {
  __shared__ float xtile[16 * 32];      // 2 KiB staging tile

  const int tid  = threadIdx.x;
  const int lane = tid & 31;
  const int wv   = tid >> 5;            // N-tile 0..3
  const int ml   = lane & 15;
  const int hh   = lane >> 4;
  const int row0 = blockIdx.x * 16;     // global row in [0, B*T)

  // async-copy assignment: each of 128 threads moves 16B of the 2KB tile
  const int arow = tid >> 3;            // 0..15
  const int aseg = tid & 7;             // 0..7 (16B segments of a 128B row)
  const float* gsrc = x + (size_t)(row0 + arow) * kDin + aseg * 4;
  const unsigned int ldst = lds_byte_off(&xtile[arow * 32 + aseg * 4]);

  v8f acc[3] = {};
  for (int kk = 0; kk < kDin; kk += 32) {
    async_load_b128(ldst, gsrc + kk);
    wait_asynccnt0();
    __syncthreads();

    // A operand from LDS: lane(ml,hh): K = hh*8+0..7 , 16+hh*8+0..7
    v4f a0 = *(const v4f*)&xtile[ml * 32 + hh * 8];
    v4f a1 = *(const v4f*)&xtile[ml * 32 + hh * 8 + 4];
    v4f a2 = *(const v4f*)&xtile[ml * 32 + 16 + hh * 8];
    v4f a3 = *(const v4f*)&xtile[ml * 32 + 16 + hh * 8 + 4];
    v16h A;
#pragma unroll
    for (int j = 0; j < 4; ++j) {
      A[j]      = (_Float16)a0[j];
      A[4 + j]  = (_Float16)a1[j];
      A[8 + j]  = (_Float16)a2[j];
      A[12 + j] = (_Float16)a3[j];
    }

    // B operands: one contiguous 32B load per matrix from the repacked image
    const _Float16* wpc =
        Wp + ((size_t)(kk >> 5) * kWpChunkStride) + ((size_t)wv * 32 + lane) * 16;
    acc[0] = wmma16(A, *(const v16h*)(wpc), acc[0]);
    acc[1] = wmma16(A, *(const v16h*)(wpc + kWpMatStride), acc[1]);
    acc[2] = wmma16(A, *(const v16h*)(wpc + 2 * kWpMatStride), acc[2]);

    __syncthreads();   // tile fully consumed before next async overwrite
  }

  const int ncol = wv * 16 + ml;
  const int bb   = row0 / kT;
  const int t0   = row0 % kT;
#pragma unroll
  for (int r = 0; r < 8; ++r) {
    q_out[(size_t)(row0 + hh * 8 + r) * kH + ncol] = (_Float16)acc[0][r];
    k_out[(size_t)(row0 + hh * 8 + r) * kH + ncol] = (_Float16)acc[1][r];
    vt_out[(size_t)bb * kH * kT + (size_t)ncol * kT + (t0 + hh * 8 + r)] =
        (_Float16)acc[2][r];
  }
}

// -------------------------------------------------------------------------
// Kernel 2: causal flash attention + dropout. One wave per (b, 16-row m-tile).
// S^T = K_tile x q^T so softmax stats are per-lane; exp'd scores pack straight
// into the A-operand layout for P x V. drop_u is a 268MB read-once stream ->
// nontemporal loads; out is write-once -> nontemporal stores.
// grid = (T/16, B), block = 32
// -------------------------------------------------------------------------
__global__ __launch_bounds__(32) void attn_kernel(
    const _Float16* __restrict__ q, const _Float16* __restrict__ k,
    const _Float16* __restrict__ vt, const float* __restrict__ drop_u,
    float* __restrict__ out) {
  const int lane = threadIdx.x & 31;
  const int ml   = lane & 15;
  const int hh   = lane >> 4;
  const int m0   = blockIdx.x * 16;
  const int bb   = blockIdx.y;
  const int mg   = m0 + ml;                      // this lane's query row

  const _Float16* qrow = q + ((size_t)bb * kT + m0 + ml) * kH;
  const v16h qB0 = *(const v16h*)(qrow + hh * 16);
  const v16h qB1 = *(const v16h*)(qrow + 32 + hh * 16);

  const _Float16* kbase = k + (size_t)bb * kT * kH;
  const _Float16* vtb   = vt + (size_t)bb * kH * kT;
  const float*    dub   = drop_u + ((size_t)bb * kT + mg) * kT;

  v8f   oacc[4] = {};
  float row_max = -__builtin_inff();
  float row_sum = 0.0f;

  for (int s0 = 0; s0 < m0 + 16; s0 += 32) {
    // --- scores S^T for 32 s-rows (two 16x16 subtiles), contract over H=64 ---
    v8f sc0 = {}, sc1 = {};
    {
      const _Float16* kr = kbase + (size_t)(s0 + ml) * kH;
      HV32 A0, A1;
      A0.h[0] = *(const v8h*)(kr + hh * 16);
      A0.h[1] = *(const v8h*)(kr + hh * 16 + 8);
      A1.h[0] = *(const v8h*)(kr + 32 + hh * 16);
      A1.h[1] = *(const v8h*)(kr + 32 + hh * 16 + 8);
      sc0 = wmma16(A0.v, qB0, sc0);
      sc0 = wmma16(A1.v, qB1, sc0);
    }
    {
      const _Float16* kr = kbase + (size_t)(s0 + 16 + ml) * kH;
      HV32 A0, A1;
      A0.h[0] = *(const v8h*)(kr + hh * 16);
      A0.h[1] = *(const v8h*)(kr + hh * 16 + 8);
      A1.h[0] = *(const v8h*)(kr + 32 + hh * 16);
      A1.h[1] = *(const v8h*)(kr + 32 + hh * 16 + 8);
      sc1 = wmma16(A0.v, qB0, sc1);
      sc1 = wmma16(A1.v, qB1, sc1);
    }

    // --- scale + causal mask; chunk row-max (per lane, then cross-half) ---
    float lmax = row_max;
#pragma unroll
    for (int r = 0; r < 8; ++r) {
      float x0 = sc0[r] * kSm;
      float x1 = sc1[r] * kSm;
      const int sg0 = s0 + hh * 8 + r;
      const int sg1 = s0 + 16 + hh * 8 + r;
      if (sg0 > mg) x0 = -__builtin_inff();
      if (sg1 > mg) x1 = -__builtin_inff();
      sc0[r] = x0;
      sc1[r] = x1;
      lmax = fmaxf(lmax, fmaxf(x0, x1));
    }
    lmax = fmaxf(lmax, __shfl_xor(lmax, 16, 32));
    const float alpha = __expf(row_max - lmax);
    row_max = lmax;

    // --- exp + normalizer sum (pre-dropout, like the reference) ---
    float lsum = 0.0f;
#pragma unroll
    for (int r = 0; r < 8; ++r) {
      const float p0 = __expf(sc0[r] - lmax);
      const float p1 = __expf(sc1[r] - lmax);
      sc0[r] = p0;
      sc1[r] = p1;
      lsum += p0 + p1;
    }
    lsum += __shfl_xor(lsum, 16, 32);
    row_sum = row_sum * alpha + lsum;

    // --- dropout on attention numerators (nontemporal 268MB stream) ---
    {
      v4f u0 = __builtin_nontemporal_load((const v4f*)(dub + s0 + hh * 8));
      v4f u1 = __builtin_nontemporal_load((const v4f*)(dub + s0 + hh * 8 + 4));
      v4f u2 = __builtin_nontemporal_load((const v4f*)(dub + s0 + 16 + hh * 8));
      v4f u3 = __builtin_nontemporal_load((const v4f*)(dub + s0 + 16 + hh * 8 + 4));
#pragma unroll
      for (int j = 0; j < 4; ++j) {
        sc0[j]     *= (u0[j] >= kDropP) ? kKeep : 0.0f;
        sc0[4 + j] *= (u1[j] >= kDropP) ? kKeep : 0.0f;
        sc1[j]     *= (u2[j] >= kDropP) ? kKeep : 0.0f;
        sc1[4 + j] *= (u3[j] >= kDropP) ? kKeep : 0.0f;
      }
    }

    // --- rescale accumulators: broadcast alpha[m] into C/D layout ---
    float av[8];
#pragma unroll
    for (int r = 0; r < 8; ++r) av[r] = __shfl(alpha, hh * 8 + r, 32);
#pragma unroll
    for (int nt = 0; nt < 4; ++nt)
#pragma unroll
      for (int r = 0; r < 8; ++r) oacc[nt][r] *= av[r];

    // --- P (S^T subtiles) packs directly into A-operand layout ---
    v16h Pa;
#pragma unroll
    for (int e = 0; e < 8; ++e) {
      Pa[e]     = (_Float16)sc0[e];   // K = hh*8 + e
      Pa[8 + e] = (_Float16)sc1[e];   // K = 16 + hh*8 + e
    }
#pragma unroll
    for (int nt = 0; nt < 4; ++nt) {
      const _Float16* vr = vtb + (size_t)(nt * 16 + ml) * kT;
      HV32 Vb;   // B operand, K = hh*8 + e, 16 + hh*8 + e
      Vb.h[0] = *(const v8h*)(vr + s0 + hh * 8);
      Vb.h[1] = *(const v8h*)(vr + s0 + 16 + hh * 8);
      oacc[nt] = wmma16(Pa, Vb.v, oacc[nt]);
    }
  }

  // --- finalize: divide by row_sum (broadcast into C/D layout), store f32 ---
  float inv[8];
#pragma unroll
  for (int r = 0; r < 8; ++r) {
    const float s = __shfl(row_sum, hh * 8 + r, 32);
    inv[r] = 1.0f / s;
  }
  float* orow = out + ((size_t)bb * kT + m0) * kH;
#pragma unroll
  for (int nt = 0; nt < 4; ++nt)
#pragma unroll
    for (int r = 0; r < 8; ++r)
      __builtin_nontemporal_store(oacc[nt][r] * inv[r],
          &orow[(size_t)(hh * 8 + r) * kH + nt * 16 + ml]);
}

// -------------------------------------------------------------------------
extern "C" void kernel_launch(void* const* d_in, const int* in_sizes, int n_in,
                              void* d_out, int out_size, void* d_ws,
                              size_t ws_size, hipStream_t stream) {
  (void)in_sizes; (void)n_in; (void)out_size; (void)ws_size;
  const float* x  = (const float*)d_in[0];
  const float* Wq = (const float*)d_in[1];
  const float* Wk = (const float*)d_in[2];
  const float* Wv = (const float*)d_in[3];
  const float* du = (const float*)d_in[4];
  float* out = (float*)d_out;

  const size_t qkv_elems = (size_t)kB * kT * kH;   // 2Mi elems -> 4 MiB f16
  _Float16* q_ws  = (_Float16*)d_ws;
  _Float16* k_ws  = q_ws + qkv_elems;
  _Float16* vt_ws = k_ws + qkv_elems;
  _Float16* wp_ws = vt_ws + qkv_elems;             // 32*12*32*16 = 196608 elems

  dim3 g0(32, 12);
  w_repack_kernel<<<g0, 32, 0, stream>>>(Wq, Wk, Wv, wp_ws);

  dim3 g1((kB * kT) / 16);
  qkv_proj_kernel<<<g1, 128, 0, stream>>>(x, wp_ws, q_ws, k_ws, vt_ws);

  dim3 g2(kT / 16, kB);
  attn_kernel<<<g2, 32, 0, stream>>>(q_ws, k_ws, vt_ws, du, out);
}
